// VerySimplifiedGNN_61272003444816
// MI455X (gfx1250) — matrix operand order repaired and checked by reading, not compiled, not run on hardware
//
#include <hip/hip_runtime.h>

typedef __attribute__((ext_vector_type(16))) _Float16 v16h;
typedef __attribute__((ext_vector_type(8)))  float    v8f;

#define F_IN  128
#define F_HID 16

// ---------------------------------------------------------------- zero fill
__global__ void gcn_fill_zero(float* __restrict__ p, long long n) {
  long long i = (long long)blockIdx.x * blockDim.x + threadIdx.x;
  if (i < n) p[i] = 0.0f;
}

// ---------------------------------------------------------------- in-degree
__global__ void gcn_degree(const long long* __restrict__ dst, int E,
                           unsigned* __restrict__ deg) {
  int e = blockIdx.x * blockDim.x + threadIdx.x;
  if (e < E) atomicAdd(&deg[(int)dst[e]], 1u);
}

// ------------------------------------------- WMMA GEMM: g = (x @ W1^T) * dinv
// One wave per 16-node tile. A = x tile (16x32 f16 per K-chunk),
// B = W1^T chunk (32x16 f16), C accumulates f32 across 4 chunks (K=128).
__global__ void gcn_gemm_scale(const float* __restrict__ x,
                               const float* __restrict__ W1,
                               const unsigned* __restrict__ deg,
                               float* __restrict__ g,
                               int N, int ntiles) {
  const int wave = (blockIdx.x * blockDim.x + threadIdx.x) >> 5;
  const int lane = threadIdx.x & 31;
  if (wave >= ntiles) return;              // wave-uniform: EXEC stays all-1s

  const int tile = wave;
  const int mrow = lane & 15;              // matrix row within tile (A load)
  const int half = lane >> 4;              // K-half selector

  // clamp row for tail-tile loads; stores are predicated below
  int arow = tile * 16 + mrow;
  if (arow > N - 1) arow = N - 1;
  const float* __restrict__ xr = x  + (long long)arow * F_IN;
  const float* __restrict__ wr = W1 + (long long)mrow * F_IN;   // B col = lane&15

  v8f c = {};
#pragma unroll
  for (int kb = 0; kb < F_IN; kb += 32) {
    // ---- A: a[0..7] = x[row][kb + 8*half + j], a[8..15] = x[row][kb+16+8*half + j]
    v16h a;
    const float4* xa = (const float4*)(xr + kb + 8 * half);
    float4 p0 = xa[0], p1 = xa[1];
    const float4* xb = (const float4*)(xr + kb + 16 + 8 * half);
    float4 p2 = xb[0], p3 = xb[1];
    a[0]=(_Float16)p0.x; a[1]=(_Float16)p0.y; a[2]=(_Float16)p0.z; a[3]=(_Float16)p0.w;
    a[4]=(_Float16)p1.x; a[5]=(_Float16)p1.y; a[6]=(_Float16)p1.z; a[7]=(_Float16)p1.w;
    a[8]=(_Float16)p2.x; a[9]=(_Float16)p2.y; a[10]=(_Float16)p2.z; a[11]=(_Float16)p2.w;
    a[12]=(_Float16)p3.x; a[13]=(_Float16)p3.y; a[14]=(_Float16)p3.z; a[15]=(_Float16)p3.w;

    // ---- B: b[0..7] = W1[n][kb + 8*half + j], b[8..15] = W1[n][kb+16+8*half + j]   (B = W1^T, col n = lane&15)
    v16h b;
    const float4* wb = (const float4*)(wr + kb + 8 * half);
    const float4* wc = (const float4*)(wr + kb + 16 + 8 * half);
    float4 q0 = wb[0], q1 = wb[1], q2 = wc[0], q3 = wc[1];
    b[0]=(_Float16)q0.x; b[1]=(_Float16)q0.y; b[2]=(_Float16)q0.z; b[3]=(_Float16)q0.w;
    b[4]=(_Float16)q1.x; b[5]=(_Float16)q1.y; b[6]=(_Float16)q1.z; b[7]=(_Float16)q1.w;
    b[8]=(_Float16)q2.x; b[9]=(_Float16)q2.y; b[10]=(_Float16)q2.z; b[11]=(_Float16)q2.w;
    b[12]=(_Float16)q3.x; b[13]=(_Float16)q3.y; b[14]=(_Float16)q3.z; b[15]=(_Float16)q3.w;

    c = __builtin_amdgcn_wmma_f32_16x16x32_f16(
        /*neg_a=*/false, a, /*neg_b=*/false, b,
        /*c_mod=*/(short)0, c, /*reuse_a=*/false, /*reuse_b=*/false);
  }

  // ---- D layout: VGPR r -> row 8*half + r, col = lane&15. Scale by dinv(row).
  const int ncol = lane & 15;
#pragma unroll
  for (int r = 0; r < 8; ++r) {
    int row = tile * 16 + 8 * half + r;
    if (row < N) {
      float dv = __frsqrt_rn((float)(deg[row] + 1u));   // +1 self-loop
      g[(long long)row * F_HID + ncol] = c[r] * dv;
    }
  }
}

// ------------------------------------------------------- edge scatter-add
// 4 threads per edge; each gathers a float4 of g[src] and atomically adds
// into tmp[dst]. g/tmp are ~6.4MB each -> L2-resident on MI455X.
__global__ void gcn_scatter(const long long* __restrict__ src,
                            const long long* __restrict__ dst, int E,
                            const float* __restrict__ g,
                            float* __restrict__ tmp) {
  int t = blockIdx.x * blockDim.x + threadIdx.x;
  if (t >= E * 4) return;
  int e = t >> 2, q = t & 3;
  int s = (int)src[e], d = (int)dst[e];
  float4 v = ((const float4*)(g + (long long)s * F_HID))[q];
  float* tp = tmp + (long long)d * F_HID + q * 4;
  atomicAdd(tp + 0, v.x);
  atomicAdd(tp + 1, v.y);
  atomicAdd(tp + 2, v.z);
  atomicAdd(tp + 3, v.w);
}

// ----------------------------------------- finalize: norm + bias + ReLU + fc1
// agg = dinv*(tmp + g) + b1  (self-loop folded: dinv^2*h == dinv*g)
__global__ void gcn_finalize(const float* __restrict__ tmp,
                             const float* __restrict__ g,
                             const unsigned* __restrict__ deg,
                             const float* __restrict__ b1,
                             const float* __restrict__ W2,
                             const float* __restrict__ b2,
                             float* __restrict__ out, int N) {
  int n = blockIdx.x * blockDim.x + threadIdx.x;
  if (n >= N) return;
  float dv = __frsqrt_rn((float)(deg[n] + 1u));
  const float* tn = tmp + (long long)n * F_HID;
  const float* gn = g   + (long long)n * F_HID;
  float a0 = 0.0f, a1 = 0.0f;
#pragma unroll
  for (int f = 0; f < F_HID; ++f) {
    float v = fmaf(dv, tn[f] + gn[f], b1[f]);
    v = fmaxf(v, 0.0f);
    a0 = fmaf(v, W2[f], a0);
    a1 = fmaf(v, W2[F_HID + f], a1);
  }
  out[2 * n + 0] = a0 + b2[0];
  out[2 * n + 1] = a1 + b2[1];
}

// ---------------------------------------------------------------- launcher
extern "C" void kernel_launch(void* const* d_in, const int* in_sizes, int n_in,
                              void* d_out, int out_size, void* d_ws, size_t ws_size,
                              hipStream_t stream) {
  const float*     x  = (const float*)d_in[0];
  const long long* ei = (const long long*)d_in[1];   // int64 [2, E]
  const float*     W1 = (const float*)d_in[2];       // [16, 128]
  const float*     b1 = (const float*)d_in[3];       // [16]
  const float*     W2 = (const float*)d_in[4];       // [2, 16]
  const float*     b2 = (const float*)d_in[5];       // [2]
  float*           out = (float*)d_out;

  const int N = in_sizes[0] / F_IN;    // 100000
  const int E = in_sizes[1] / 2;       // 3200000
  const long long* src = ei;
  const long long* dst = ei + E;

  // workspace: g[N*16] | tmp[N*16] | deg[N]  (~13.2 MB)
  float*    g   = (float*)d_ws;
  float*    tmp = g + (long long)N * F_HID;
  unsigned* deg = (unsigned*)(tmp + (long long)N * F_HID);

  // 1) zero tmp + deg (contiguous; zero bit pattern shared)
  {
    long long nz = (long long)N * F_HID + N;
    int blocks = (int)((nz + 255) / 256);
    gcn_fill_zero<<<blocks, 256, 0, stream>>>(tmp, nz);
  }
  // 2) in-degrees
  gcn_degree<<<(E + 255) / 256, 256, 0, stream>>>(dst, E, deg);
  // 3) WMMA GEMM + normalize -> g
  {
    int ntiles = (N + 15) / 16;
    int blocks = (ntiles + 7) / 8;            // 8 waves (256 thr) per block
    gcn_gemm_scale<<<blocks, 256, 0, stream>>>(x, W1, deg, g, N, ntiles);
  }
  // 4) edge scatter-add
  {
    long long nt = (long long)E * 4;
    gcn_scatter<<<(int)((nt + 255) / 256), 256, 0, stream>>>(src, dst, E, g, tmp);
  }
  // 5) finalize + fc1
  gcn_finalize<<<(N + 255) / 256, 256, 0, stream>>>(tmp, g, deg, b1, W2, b2, out, N);
}
